// LinearAttention_69346541961569
// MI455X (gfx1250) — matrix operand order repaired and checked by reading, not compiled, not run on hardware
//
#include <hip/hip_runtime.h>
#include <math.h>
#include <stdint.h>

typedef _Float16 half_t;
typedef __attribute__((ext_vector_type(16))) _Float16 v16h;
typedef __attribute__((ext_vector_type(8)))  _Float16 v8h;
typedef __attribute__((ext_vector_type(4)))  _Float16 v4h;
typedef __attribute__((ext_vector_type(8)))  float    v8f;

#define BS_    2
#define SEQ_   4096
#define DIM_   4096
#define NH_    32
#define NKV_   8
#define HD_    128
#define NCHUNK_ 32

// ---------------------------------------------------------------------------
// WMMA helpers (CDNA5 wave32, 16x16x32 f16 -> f32)
// ---------------------------------------------------------------------------
__device__ __forceinline__ v8f wmma_f16(v16h a, v16h b, v8f c) {
  return __builtin_amdgcn_wmma_f32_16x16x32_f16(false, a, false, b, (short)0, c,
                                                false, false);
}

// A fragment: 16x32 f16, row-major tile in LDS with leading-dim ldh (halves).
// Lane L (<16) holds row M=L, K {0..7,16..23}; lane L>=16 holds K {8..15,24..31}.
__device__ __forceinline__ v16h load_a_frag(const half_t* tile, int ldh, int lane) {
  int row = lane & 15;
  int k   = (lane >> 4) * 8;
  const half_t* p = tile + row * ldh + k;
  v8h lo = *(const v8h*)(p);
  v8h hi = *(const v8h*)(p + 16);
  v16h r;
#pragma unroll
  for (int i = 0; i < 8; ++i) { r[i] = lo[i]; r[i + 8] = hi[i]; }
  return r;
}

// B fragment: 32x16 f16. LDS tile stored transposed [n][k] with leading-dim ldh.
// Lane L holds column N=L&15; lane L<16 holds K {0..7,16..23}; lane L>=16 holds K {8..15,24..31}.
__device__ __forceinline__ v16h load_b_frag(const half_t* tileT, int ldh, int lane) {
  int n = lane & 15;
  int k = (lane >> 4) * 8;
  const half_t* p = tileT + n * ldh + k;
  v8h lo = *(const v8h*)(p);
  v8h hi = *(const v8h*)(p + 16);
  v16h r;
#pragma unroll
  for (int i = 0; i < 8; ++i) { r[i] = lo[i]; r[i + 8] = hi[i]; }
  return r;
}

__device__ __forceinline__ void cvt4(const float* p, half_t* dst) {
  float4 v = *(const float4*)p;
  dst[0] = (half_t)v.x; dst[1] = (half_t)v.y;
  dst[2] = (half_t)v.z; dst[3] = (half_t)v.w;
}
__device__ __forceinline__ void cvt4(const half_t* p, half_t* dst) {
  v4h v = *(const v4h*)p;
  dst[0] = v[0]; dst[1] = v[1]; dst[2] = v[2]; dst[3] = v[3];
}

// Async global->LDS 16-byte copy (CDNA5, tracked by ASYNCcnt).
__device__ __forceinline__ void async_copy_b128(const half_t* gsrc, half_t* lds_dst) {
  unsigned lds = (unsigned)(uintptr_t)lds_dst;               // LDS byte offset
  unsigned long long g = (unsigned long long)(uintptr_t)gsrc;
  asm volatile("global_load_async_to_lds_b128 %0, %1, off"
               :: "v"(lds), "v"(g) : "memory");
}
__device__ __forceinline__ void wait_asynccnt0() {
  asm volatile("s_wait_asynccnt 0x0" ::: "memory");
}

// ---------------------------------------------------------------------------
// GEMM: Y[m][n] = sum_k X[m][k] * W[n][k]   (X: MxK InT, W: NxK f32 row-major)
// Block tile 128x128, BK=64, 256 threads = 8 waves, each wave 2x4 WMMA tiles.
// When InT is f16, the A tile is staged with GLOBAL_LOAD_ASYNC_TO_LDS_B128.
// ---------------------------------------------------------------------------
template <typename InT, typename OutT>
__global__ __launch_bounds__(256) void gemm_xwt(const InT* __restrict__ X,
                                                const float* __restrict__ W,
                                                OutT* __restrict__ Y,
                                                int M, int N, int K) {
  __shared__ alignas(16) half_t As[128][72];
  __shared__ alignas(16) half_t Bs[128][72];
  const int tid  = threadIdx.x;
  const int lane = tid & 31;
  const int wave = tid >> 5;
  const int m0 = blockIdx.x * 128;
  const int n0 = blockIdx.y * 128;
  const int waveM = (wave & 3) * 32;
  const int waveN = (wave >> 2) * 64;
  v8f acc[2][4] = {};

  for (int k0 = 0; k0 < K; k0 += 64) {
    __syncthreads();
    if constexpr (sizeof(InT) == 2) {
      // A tile: pure byte copy, async DMA into padded LDS layout.
#pragma unroll
      for (int it = 0; it < 4; ++it) {
        int idx = tid + it * 256;        // 0..1023 16B chunks
        int r   = idx >> 3;              // 0..127
        int c8  = (idx & 7) * 8;         // half offset, 16B granules
        async_copy_b128((const half_t*)X + (size_t)(m0 + r) * K + k0 + c8,
                        &As[r][c8]);
      }
#pragma unroll
      for (int it = 0; it < 8; ++it) {
        int idx = tid + it * 256;
        int r   = idx >> 4;
        int c4  = (idx & 15) * 4;
        cvt4(W + (size_t)(n0 + r) * K + k0 + c4, &Bs[r][c4]);
      }
      wait_asynccnt0();
    } else {
#pragma unroll
      for (int it = 0; it < 8; ++it) {
        int idx = tid + it * 256;         // 0..2047
        int r   = idx >> 4;               // 0..127
        int c4  = (idx & 15) * 4;         // 0..60
        cvt4(X + (size_t)(m0 + r) * K + k0 + c4, &As[r][c4]);
        cvt4(W + (size_t)(n0 + r) * K + k0 + c4, &Bs[r][c4]);
      }
    }
    __syncthreads();
#pragma unroll
    for (int kk = 0; kk < 64; kk += 32) {
      v16h a0 = load_a_frag(&As[waveM][kk],      72, lane);
      v16h a1 = load_a_frag(&As[waveM + 16][kk], 72, lane);
      v16h b0 = load_b_frag(&Bs[waveN][kk],      72, lane);
      v16h b1 = load_b_frag(&Bs[waveN + 16][kk], 72, lane);
      v16h b2 = load_b_frag(&Bs[waveN + 32][kk], 72, lane);
      v16h b3 = load_b_frag(&Bs[waveN + 48][kk], 72, lane);
      acc[0][0] = wmma_f16(a0, b0, acc[0][0]);
      acc[0][1] = wmma_f16(a0, b1, acc[0][1]);
      acc[0][2] = wmma_f16(a0, b2, acc[0][2]);
      acc[0][3] = wmma_f16(a0, b3, acc[0][3]);
      acc[1][0] = wmma_f16(a1, b0, acc[1][0]);
      acc[1][1] = wmma_f16(a1, b1, acc[1][1]);
      acc[1][2] = wmma_f16(a1, b2, acc[1][2]);
      acc[1][3] = wmma_f16(a1, b3, acc[1][3]);
    }
  }
  const int colB = n0 + waveN + (lane & 15);
  const int rowB = m0 + waveM + ((lane >> 4) * 8);
#pragma unroll
  for (int i = 0; i < 2; ++i)
#pragma unroll
    for (int j = 0; j < 4; ++j)
#pragma unroll
      for (int rr = 0; rr < 8; ++rr)
        Y[(size_t)(rowB + i * 16 + rr) * N + colB + j * 16] = (OutT)acc[i][j][rr];
}

// ---------------------------------------------------------------------------
// Rotary embedding, in place, on f16 [b][s][heads][128] (pairs along d).
// ---------------------------------------------------------------------------
__global__ void rope_kernel(half_t* __restrict__ t, const float* __restrict__ fc,
                            int heads, int totalPairs) {
  int p = blockIdx.x * blockDim.x + threadIdx.x;
  if (p >= totalPairs) return;
  int i = p & 63;
  int s = ((p >> 6) / heads) % SEQ_;
  float co = fc[(s * 64 + i) * 2 + 0];
  float si = fc[(s * 64 + i) * 2 + 1];
  half_t* q = t + (size_t)p * 2;
  float t0 = (float)q[0], t1 = (float)q[1];
  q[0] = (half_t)(t0 * co - t1 * si);
  q[1] = (half_t)(t0 * si + t1 * co);
}

// ---------------------------------------------------------------------------
// Per-(b, kvh, chunk): m_loc[d]=max_r k, e=exp(k-m), A_loc[d]=sum_r e,
// B_loc[d][e'] = sum_r e[r][d] * v[r][e']   (WMMA, K=r split into 2 halves)
// ---------------------------------------------------------------------------
__global__ __launch_bounds__(256) void chunk_stats_kernel(
    const half_t* __restrict__ xk, const half_t* __restrict__ xv,
    float* __restrict__ mloc, float* __restrict__ aloc,
    float* __restrict__ bloc) {
  __shared__ alignas(16) half_t Et[128][72];   // [d][r_local]  (A: M=d, K=r)
  __shared__ alignas(16) half_t Vt[128][72];   // [e'][r_local] (B^T)
  const int c = blockIdx.x, kvh = blockIdx.y, b = blockIdx.z;
  const int tid = threadIdx.x, lane = tid & 31, wave = tid >> 5;
  const int rowStride = NKV_ * HD_;            // 1024 halves per seq row
  const half_t* kb = xk + (size_t)(b * SEQ_ + c * 128) * rowStride + kvh * HD_;
  const half_t* vb = xv + (size_t)(b * SEQ_ + c * 128) * rowStride + kvh * HD_;

  float mx = -INFINITY, asum = 0.f;
  if (tid < 128) {
    for (int r = 0; r < 128; ++r)
      mx = fmaxf(mx, (float)kb[(size_t)r * rowStride + tid]);
  }
  const int waveM = (wave & 3) * 32;
  const int waveN = (wave >> 2) * 64;
  v8f acc[2][4] = {};

  for (int hK = 0; hK < 2; ++hK) {
    __syncthreads();
    if (tid < 128) {
      for (int rr = 0; rr < 64; ++rr) {
        float e = __expf((float)kb[(size_t)(hK * 64 + rr) * rowStride + tid] - mx);
        asum += e;
        Et[tid][rr] = (half_t)e;
      }
    }
    for (int idx = tid; idx < 128 * 64; idx += 256) {
      int rr = idx >> 7, e = idx & 127;
      Vt[e][rr] = vb[(size_t)(hK * 64 + rr) * rowStride + e];
    }
    __syncthreads();
#pragma unroll
    for (int kk = 0; kk < 64; kk += 32) {
      v16h a0 = load_a_frag(&Et[waveM][kk],      72, lane);
      v16h a1 = load_a_frag(&Et[waveM + 16][kk], 72, lane);
      v16h b0 = load_b_frag(&Vt[waveN][kk],      72, lane);
      v16h b1 = load_b_frag(&Vt[waveN + 16][kk], 72, lane);
      v16h b2 = load_b_frag(&Vt[waveN + 32][kk], 72, lane);
      v16h b3 = load_b_frag(&Vt[waveN + 48][kk], 72, lane);
      acc[0][0] = wmma_f16(a0, b0, acc[0][0]);
      acc[0][1] = wmma_f16(a0, b1, acc[0][1]);
      acc[0][2] = wmma_f16(a0, b2, acc[0][2]);
      acc[0][3] = wmma_f16(a0, b3, acc[0][3]);
      acc[1][0] = wmma_f16(a1, b0, acc[1][0]);
      acc[1][1] = wmma_f16(a1, b1, acc[1][1]);
      acc[1][2] = wmma_f16(a1, b2, acc[1][2]);
      acc[1][3] = wmma_f16(a1, b3, acc[1][3]);
    }
  }
  if (tid < 128) {
    size_t o = ((size_t)(b * NKV_ + kvh) * NCHUNK_ + c) * 128 + tid;
    mloc[o] = mx;
    aloc[o] = asum;
  }
  size_t bo = (((size_t)(b * NKV_ + kvh) * NCHUNK_ + c) * 128) * 128;
  const int colB = waveN + (lane & 15);
  const int rowB = waveM + ((lane >> 4) * 8);
#pragma unroll
  for (int i = 0; i < 2; ++i)
#pragma unroll
    for (int j = 0; j < 4; ++j)
#pragma unroll
      for (int rr = 0; rr < 8; ++rr)
        bloc[bo + (size_t)(rowB + i * 16 + rr) * 128 + colB + j * 16] =
            acc[i][j][rr];
}

// ---------------------------------------------------------------------------
// Sequential scan over chunks; running (m, A, B) per (b,kvh). kv = B/A.
// Thread t owns d = t/2, e range [(t&1)*64, +64).
// ---------------------------------------------------------------------------
__global__ __launch_bounds__(256) void scan_kernel(
    const float* __restrict__ mloc, const float* __restrict__ aloc,
    const float* __restrict__ bloc, half_t* __restrict__ kv) {
  const int bk = blockIdx.x;   // b*NKV + kvh
  const int tid = threadIdx.x;
  const int d  = tid >> 1;
  const int e0 = (tid & 1) * 64;
  float m = -INFINITY, A = 0.f;
  float B[64];
#pragma unroll
  for (int i = 0; i < 64; ++i) B[i] = 0.f;
  for (int c = 0; c < NCHUNK_; ++c) {
    size_t o = ((size_t)bk * NCHUNK_ + c) * 128 + d;
    float ml = mloc[o], Al = aloc[o];
    float mn = fmaxf(m, ml);
    float so = __expf(m - mn);     // first iter: exp(-inf)=0
    float sn = __expf(ml - mn);
    A = A * so + Al * sn;
    m = mn;
    const float* bl = bloc + o * 128 + e0;
    half_t* op = kv + o * 128 + e0;
    float inv = 1.f / A;
#pragma unroll
    for (int i = 0; i < 64; ++i) {
      B[i] = B[i] * so + bl[i] * sn;
      op[i] = (half_t)(B[i] * inv);
    }
  }
}

// ---------------------------------------------------------------------------
// Per-(b, h, chunk): out[r][e] = softmax_d(q[r][:]) @ kv[d][e]  (WMMA)
// ---------------------------------------------------------------------------
__global__ __launch_bounds__(256) void attn_out_kernel(
    const half_t* __restrict__ xq, const half_t* __restrict__ kv,
    half_t* __restrict__ X2) {
  __shared__ alignas(16) half_t Qs[128][72];    // [r][d_local]
  __shared__ alignas(16) half_t KVt[128][72];   // [e][d_local]
  const int c = blockIdx.x, h = blockIdx.y, b = blockIdx.z;
  const int kvh = h >> 2;   // n_rep = 4
  const int tid = threadIdx.x, lane = tid & 31, wave = tid >> 5;
  const int row = tid >> 1, part = tid & 1;

  const half_t* qrow = xq + (size_t)(b * SEQ_ + c * 128 + row) * (NH_ * HD_) +
                       h * HD_ + part * 64;
  float v[64];
  float mx = -INFINITY;
#pragma unroll
  for (int i = 0; i < 64; ++i) { v[i] = (float)qrow[i]; mx = fmaxf(mx, v[i]); }
  mx = fmaxf(mx, __shfl_xor(mx, 1, 32));
  float ssum = 0.f;
#pragma unroll
  for (int i = 0; i < 64; ++i) { v[i] = __expf(v[i] - mx); ssum += v[i]; }
  ssum += __shfl_xor(ssum, 1, 32);
  float inv = 1.f / ssum;
#pragma unroll
  for (int i = 0; i < 64; ++i) v[i] *= inv;

  const half_t* kvb = kv + (((size_t)(b * NKV_ + kvh) * NCHUNK_ + c) * 128) * 128;
  const int waveM = (wave & 3) * 32;
  const int waveN = (wave >> 2) * 64;
  v8f acc[2][4] = {};

  for (int hK = 0; hK < 2; ++hK) {
    __syncthreads();
    if (part == hK) {
#pragma unroll
      for (int i = 0; i < 64; ++i) Qs[row][i] = (half_t)v[i];
    }
    for (int idx = tid; idx < 128 * 64; idx += 256) {
      int dd = idx >> 7, e = idx & 127;
      KVt[e][dd] = kvb[(size_t)(hK * 64 + dd) * 128 + e];
    }
    __syncthreads();
#pragma unroll
    for (int kk = 0; kk < 64; kk += 32) {
      v16h a0 = load_a_frag(&Qs[waveM][kk],      72, lane);
      v16h a1 = load_a_frag(&Qs[waveM + 16][kk], 72, lane);
      v16h b0 = load_b_frag(&KVt[waveN][kk],      72, lane);
      v16h b1 = load_b_frag(&KVt[waveN + 16][kk], 72, lane);
      v16h b2 = load_b_frag(&KVt[waveN + 32][kk], 72, lane);
      v16h b3 = load_b_frag(&KVt[waveN + 48][kk], 72, lane);
      acc[0][0] = wmma_f16(a0, b0, acc[0][0]);
      acc[0][1] = wmma_f16(a0, b1, acc[0][1]);
      acc[0][2] = wmma_f16(a0, b2, acc[0][2]);
      acc[0][3] = wmma_f16(a0, b3, acc[0][3]);
      acc[1][0] = wmma_f16(a1, b0, acc[1][0]);
      acc[1][1] = wmma_f16(a1, b1, acc[1][1]);
      acc[1][2] = wmma_f16(a1, b2, acc[1][2]);
      acc[1][3] = wmma_f16(a1, b3, acc[1][3]);
    }
  }
  half_t* ob = X2 + (size_t)(b * SEQ_ + c * 128) * (NH_ * HD_) + h * HD_;
  const int colB = waveN + (lane & 15);
  const int rowB = waveM + ((lane >> 4) * 8);
#pragma unroll
  for (int i = 0; i < 2; ++i)
#pragma unroll
    for (int j = 0; j < 4; ++j)
#pragma unroll
      for (int rr = 0; rr < 8; ++rr)
        ob[(size_t)(rowB + i * 16 + rr) * (NH_ * HD_) + colB + j * 16] =
            (half_t)acc[i][j][rr];
}

// ---------------------------------------------------------------------------
extern "C" void kernel_launch(void* const* d_in, const int* in_sizes, int n_in,
                              void* d_out, int out_size, void* d_ws, size_t ws_size,
                              hipStream_t stream) {
  (void)in_sizes; (void)n_in; (void)out_size; (void)ws_size;
  const float* x  = (const float*)d_in[0];
  const float* fc = (const float*)d_in[1];
  const float* wq = (const float*)d_in[2];
  const float* wk = (const float*)d_in[3];
  const float* wv = (const float*)d_in[4];
  const float* wo = (const float*)d_in[5];
  float* out = (float*)d_out;

  const int M = BS_ * SEQ_;                       // 8192
  char* ws = (char*)d_ws;
  size_t off = 0;
  auto take = [&](size_t bytes) { void* p = ws + off; off += (bytes + 255) & ~(size_t)255; return p; };

  half_t* xq   = (half_t*)take((size_t)M * (NH_ * HD_) * 2);              // 64 MB
  half_t* xk   = (half_t*)take((size_t)M * (NKV_ * HD_) * 2);             // 16 MB
  half_t* xv   = (half_t*)take((size_t)M * (NKV_ * HD_) * 2);             // 16 MB
  half_t* X2   = (half_t*)take((size_t)M * (NH_ * HD_) * 2);              // 64 MB
  half_t* kvb  = (half_t*)take((size_t)BS_ * NKV_ * NCHUNK_ * 128 * 128 * 2); // 16 MB
  float*  bloc = (float*) take((size_t)BS_ * NKV_ * NCHUNK_ * 128 * 128 * 4); // 32 MB
  float*  mloc = (float*) take((size_t)BS_ * NKV_ * NCHUNK_ * 128 * 4);
  float*  aloc = (float*) take((size_t)BS_ * NKV_ * NCHUNK_ * 128 * 4);

  dim3 blk(256);
  // 1) QKV projections (f32 x, f32 W -> f16)
  gemm_xwt<float, half_t><<<dim3(M / 128, 32), blk, 0, stream>>>(x, wq, xq, M, NH_ * HD_, DIM_);
  gemm_xwt<float, half_t><<<dim3(M / 128, 8),  blk, 0, stream>>>(x, wk, xk, M, NKV_ * HD_, DIM_);
  gemm_xwt<float, half_t><<<dim3(M / 128, 8),  blk, 0, stream>>>(x, wv, xv, M, NKV_ * HD_, DIM_);
  // 2) rotary
  int qPairs = M * NH_ * 64;
  int kPairs = M * NKV_ * 64;
  rope_kernel<<<qPairs / 256, blk, 0, stream>>>(xq, fc, NH_,  qPairs);
  rope_kernel<<<kPairs / 256, blk, 0, stream>>>(xk, fc, NKV_, kPairs);
  // 3) per-chunk stats + e^T @ v
  chunk_stats_kernel<<<dim3(NCHUNK_, NKV_, BS_), blk, 0, stream>>>(xk, xv, mloc, aloc, bloc);
  // 4) online rescaling scan -> kv
  scan_kernel<<<BS_ * NKV_, blk, 0, stream>>>(mloc, aloc, bloc, kvb);
  // 5) softmax(q) @ kv
  attn_out_kernel<<<dim3(NCHUNK_, NH_, BS_), blk, 0, stream>>>(xq, kvb, X2);
  // 6) output projection (f16 X2 via async-to-LDS, f32 wo -> f32 out)
  gemm_xwt<half_t, float><<<dim3(M / 128, 32), blk, 0, stream>>>(X2, wo, out, M, DIM_, DIM_);
}
